// MultiHeadAttention_14903536517146
// MI455X (gfx1250) — matrix operand order repaired and checked by reading, not compiled, not run on hardware
//
#include <hip/hip_runtime.h>

// MI455X / gfx1250 — wave32, WMMA 16x16x32 f16->f32, async global->LDS staging.
// MHA forward: B=8,S=1024,D=1024,H=16,HD=64. fp32 in/out, f16 tensor-core math.
// Workspace usage: 88 MB of d_ws (layout below).

typedef _Float16 f16;
typedef _Float16 v16h __attribute__((ext_vector_type(16)));
typedef _Float16 v4h  __attribute__((ext_vector_type(4)));
typedef float    v8f  __attribute__((ext_vector_type(8)));
typedef unsigned int v4u __attribute__((ext_vector_type(4)));

#define Dd 1024
#define Hh 16
#define HDc 64
#define Ss 1024
#define Bb 8
#define BS (Bb*Ss)

union F16Frag { v16h v; v4u u[2]; };

__device__ __forceinline__ v8f zero8() {
  v8f z = {0.f,0.f,0.f,0.f,0.f,0.f,0.f,0.f};
  return z;
}

__device__ __forceinline__ v8f wmma32(v16h a, v16h b, v8f c) {
  // D = A(16x32 f16) * B(32x16 f16) + C(16x16 f32)
  return __builtin_amdgcn_wmma_f32_16x16x32_f16(false, a, false, b, (short)0, c,
                                                false, false);
}

// CDNA5 waits not exposed as builtins on this toolchain -> inline asm.
__device__ __forceinline__ void wait_dscnt0() {
  asm volatile("s_wait_dscnt 0" ::: "memory");
}
__device__ __forceinline__ void wait_asynccnt0() {
  asm volatile("s_wait_asynccnt 0" ::: "memory");
}

// Async DMA: 16B per lane, global memory -> LDS, tracked by ASYNCcnt (no VGPR
// data round-trip). ldsoff = per-lane LDS byte offset (low 32 bits of generic
// LDS address), gaddr = per-lane 64-bit global address.
__device__ __forceinline__ void async_g2lds_b128(unsigned ldsoff, const void* gaddr) {
  asm volatile("global_load_async_to_lds_b128 %0, %1, off"
               :: "v"(ldsoff), "v"(gaddr) : "memory");
}

// ---------------------------------------------------------------- fp32 -> f16
__global__ __launch_bounds__(256) void mha_cvt(const float* __restrict__ in,
                                               f16* __restrict__ out, int n4) {
  int i = blockIdx.x * 256 + threadIdx.x;
  if (i >= n4) return;
  float4 v = ((const float4*)in)[i];
  v4h o;
  o.x = (f16)v.x; o.y = (f16)v.y; o.z = (f16)v.z; o.w = (f16)v.w;
  *(v4h*)(out + 4 * (size_t)i) = o;
}

// ----------------------------------------------------- QKV projection (WMMA)
// C[m,n] = sum_d X[m,d]*W[n,d] + bias[n]; wave computes a 16x64 strip.
// mode 0: Q -> [b,h,s,hd] scaled by 0.125   mode 1: K -> [b,h,s,hd]
// mode 2: V -> [b,h,hd,s] (transposed, for contiguous P@V B-frags)
__global__ __launch_bounds__(128) void mha_qkv_gemm(
    const f16* __restrict__ X, const f16* __restrict__ W,
    const float* __restrict__ bias, f16* __restrict__ out, const int mode) {
  const int lane = threadIdx.x & 31;
  const int wave = threadIdx.x >> 5;
  const int lm   = lane & 15;
  const int hlf  = lane >> 4;
  const int m0 = blockIdx.x * 16;
  const int n0 = (blockIdx.y * 4 + wave) * 64;

  // A layout: lane=M, halves = K=k0+16*hlf+0..15 (contiguous, matches B)
  const f16* arow = X + (size_t)(m0 + lm) * Dd + 16 * hlf;
  // B layout: lane=N, halves = K=k0+16*hlf+0..15 (one contiguous 32B read)
  const f16* brow = W + (size_t)(n0 + lm) * Dd + 16 * hlf;

  v8f acc[4];
#pragma unroll
  for (int t = 0; t < 4; ++t) acc[t] = zero8();

  F16Frag a, bf[4];
  a.u[0] = *(const v4u*)(arow);
  a.u[1] = *(const v4u*)(arow + 8);
#pragma unroll
  for (int t = 0; t < 4; ++t) {
    const f16* p = brow + (size_t)t * 16 * Dd;
    bf[t].u[0] = *(const v4u*)(p);
    bf[t].u[1] = *(const v4u*)(p + 8);
  }

  for (int k0 = 32; k0 < Dd; k0 += 32) {
    F16Frag a2, b2[4];
    a2.u[0] = *(const v4u*)(arow + k0);
    a2.u[1] = *(const v4u*)(arow + k0 + 8);
#pragma unroll
    for (int t = 0; t < 4; ++t) {
      const f16* p = brow + (size_t)t * 16 * Dd + k0;
      b2[t].u[0] = *(const v4u*)(p);
      b2[t].u[1] = *(const v4u*)(p + 8);
    }
#pragma unroll
    for (int t = 0; t < 4; ++t) acc[t] = wmma32(a.v, bf[t].v, acc[t]);
    a = a2;
#pragma unroll
    for (int t = 0; t < 4; ++t) bf[t] = b2[t];
  }
#pragma unroll
  for (int t = 0; t < 4; ++t) acc[t] = wmma32(a.v, bf[t].v, acc[t]);

#pragma unroll
  for (int t = 0; t < 4; ++t) {
    const int n  = n0 + t * 16 + lm;
    const float bv = bias[n];
    const int hh = n >> 6;
    const int hd = n & 63;
#pragma unroll
    for (int j = 0; j < 8; ++j) {
      const int row = m0 + j + 8 * hlf;  // C layout: M = j + 8*half, N = lm
      const int bb  = row >> 10;
      const int s   = row & 1023;
      float v = acc[t][j] + bv;
      if (mode == 0)
        out[(((size_t)(bb * Hh + hh)) * Ss + s) * HDc + hd] = (f16)(v * 0.125f);
      else if (mode == 1)
        out[(((size_t)(bb * Hh + hh)) * Ss + s) * HDc + hd] = (f16)v;
      else
        out[(((size_t)(bb * Hh + hh)) * HDc + hd) * Ss + s] = (f16)v;
    }
  }
}

// ------------------------------------------------- flash attention per (b,h)
// Block = (b,h, quarter-of-S). K[S,HD] and Vt[HD,S] for the head are staged
// into LDS with GLOBAL_LOAD_ASYNC_TO_LDS_B128 (ASYNCcnt DMA path, no VGPR
// round-trip). 128KB + 128KB + 8KB P scratch = 264KB < 320KB WGP LDS.
__global__ __launch_bounds__(256) void mha_attn(
    const f16* __restrict__ Q, const f16* __restrict__ K,
    const f16* __restrict__ Vt, f16* __restrict__ Cc) {
  extern __shared__ char smem_raw[];
  f16* Ksm  = (f16*)smem_raw;       // [S][HD]
  f16* Vsm  = Ksm + Ss * HDc;       // [HD][S]
  f16* Pall = Vsm + HDc * Ss;       // 8 waves x 512 halves

  const int tid  = threadIdx.x;
  const int lane = tid & 31;
  const int wave = tid >> 5;
  const int lm   = lane & 15;
  const int hlf  = lane >> 4;
  const int qb = blockIdx.x & 3;
  const int bh = blockIdx.x >> 2;
  const int b  = bh >> 4;
  const int h  = bh & 15;

  // Async-DMA stage of K and V^T into LDS (8192 x 16B each).
  {
    const f16* gk = K  + (size_t)bh * (Ss * HDc);
    const f16* gv = Vt + (size_t)bh * (Ss * HDc);
    const unsigned koff = (unsigned)(unsigned long long)(void*)Ksm;
    const unsigned voff = (unsigned)(unsigned long long)(void*)Vsm;
    for (int i = tid; i < (Ss * HDc) / 8; i += 256) {
      async_g2lds_b128(koff + 16u * i, gk + 8 * (size_t)i);
      async_g2lds_b128(voff + 16u * i, gv + 8 * (size_t)i);
    }
    wait_asynccnt0();   // each wave drains its own ASYNCcnt ...
  }
  __syncthreads();      // ... then all waves rendezvous: LDS fully populated

  f16* Pw = Pall + wave * 512;
  const f16* Qg = Q + (size_t)bh * (Ss * HDc);

  for (int qt = qb * 16 + wave; qt < qb * 16 + 16; qt += 8) {
    const int q0 = qt * 16;
    F16Frag qa0, qa1;  // Q tile 16x64 as two A-fragments (K = hd)
    const f16* qrow = Qg + (size_t)(q0 + lm) * HDc + 16 * hlf;
    qa0.u[0] = *(const v4u*)(qrow);
    qa0.u[1] = *(const v4u*)(qrow + 8);
    qa1.u[0] = *(const v4u*)(qrow + 32);
    qa1.u[1] = *(const v4u*)(qrow + 40);

    v8f o0 = zero8(), o1 = zero8(), o2 = zero8(), o3 = zero8();
    float mrow[8], lrow[8];
#pragma unroll
    for (int j = 0; j < 8; ++j) { mrow[j] = -1e30f; lrow[j] = 0.f; }

    for (int c = 0; c < 32; ++c) {
      const int s0 = c * 32;

      // Issue ALL chunk loads up front: 8 ds_load_b128 for K-frags, then 8 for
      // V-frags. V-frag latency hides under the score WMMAs + softmax VALU.
      F16Frag kb00, kb01, kb10, kb11;
      {
        const f16* kr0 = Ksm + (size_t)(s0 + lm) * HDc + 16 * hlf;        // s-tile 0
        const f16* kr1 = kr0 + 16 * HDc;                                  // s-tile 1
        kb00.u[0] = *(const v4u*)(kr0);      kb00.u[1] = *(const v4u*)(kr0 + 8);
        kb01.u[0] = *(const v4u*)(kr0 + 32); kb01.u[1] = *(const v4u*)(kr0 + 40);
        kb10.u[0] = *(const v4u*)(kr1);      kb10.u[1] = *(const v4u*)(kr1 + 8);
        kb11.u[0] = *(const v4u*)(kr1 + 32); kb11.u[1] = *(const v4u*)(kr1 + 40);
      }
      F16Frag vb0, vb1, vb2, vb3;
      {
        const f16* vr = Vsm + (size_t)lm * Ss + s0 + 16 * hlf;
        vb0.u[0] = *(const v4u*)(vr);               vb0.u[1] = *(const v4u*)(vr + 8);
        vb1.u[0] = *(const v4u*)(vr + 16 * Ss);     vb1.u[1] = *(const v4u*)(vr + 16 * Ss + 8);
        vb2.u[0] = *(const v4u*)(vr + 32 * Ss);     vb2.u[1] = *(const v4u*)(vr + 32 * Ss + 8);
        vb3.u[0] = *(const v4u*)(vr + 48 * Ss);     vb3.u[1] = *(const v4u*)(vr + 48 * Ss + 8);
      }

      // Score tiles: S = Q(16x64) @ K^T, two chained K=32 WMMAs per tile.
      v8f sa = zero8(), sb = zero8();
      sa = wmma32(qa0.v, kb00.v, sa);
      sa = wmma32(qa1.v, kb01.v, sa);
      sb = wmma32(qa0.v, kb10.v, sb);
      sb = wmma32(qa1.v, kb11.v, sb);

      // Online softmax across the 32 new columns (row = j + 8*hlf, col = lm).
#pragma unroll
      for (int j = 0; j < 8; ++j) {
        float a0 = sa[j], a1 = sb[j];
        float mx = fmaxf(a0, a1);
        mx = fmaxf(mx, __shfl_xor(mx, 1));
        mx = fmaxf(mx, __shfl_xor(mx, 2));
        mx = fmaxf(mx, __shfl_xor(mx, 4));
        mx = fmaxf(mx, __shfl_xor(mx, 8));
        float mnew  = fmaxf(mrow[j], mx);
        float alpha = __expf(mrow[j] - mnew);
        float p0 = __expf(a0 - mnew);
        float p1 = __expf(a1 - mnew);
        float rs = p0 + p1;
        rs += __shfl_xor(rs, 1);
        rs += __shfl_xor(rs, 2);
        rs += __shfl_xor(rs, 4);
        rs += __shfl_xor(rs, 8);
        lrow[j] = lrow[j] * alpha + rs;
        mrow[j] = mnew;
        o0[j] *= alpha; o1[j] *= alpha; o2[j] *= alpha; o3[j] *= alpha;
        const int prow = j + 8 * hlf;
        Pw[prow * 32 + lm]      = (f16)p0;   // P tile row-major 16x32
        Pw[prow * 32 + 16 + lm] = (f16)p1;
      }
      wait_dscnt0();  // wave-private LDS RAW fence (stores -> cross-lane loads)

      F16Frag pa;  // re-read P in A layout (lane = M, K contiguous)
      pa.u[0] = *(const v4u*)(Pw + lm * 32 + 16 * hlf);
      pa.u[1] = *(const v4u*)(Pw + lm * 32 + 16 * hlf + 8);

      // O[16x64] += P[16x32] @ V[32x64]; V-frags already resident.
      o0 = wmma32(pa.v, vb0.v, o0);
      o1 = wmma32(pa.v, vb1.v, o1);
      o2 = wmma32(pa.v, vb2.v, o2);
      o3 = wmma32(pa.v, vb3.v, o3);
    }

    // Normalize and write head-concat layout [b, s, h*HD + hd] in f16.
#pragma unroll
    for (int j = 0; j < 8; ++j) {
      float inv = 1.0f / lrow[j];
      int row = q0 + j + 8 * hlf;
      f16* cp = Cc + ((size_t)(b * Ss + row)) * Dd + h * HDc + lm;
      cp[0]  = (f16)(o0[j] * inv);
      cp[16] = (f16)(o1[j] * inv);
      cp[32] = (f16)(o2[j] * inv);
      cp[48] = (f16)(o3[j] * inv);
    }
  }
}

// ------------------------------------------------ output projection (fp32 out)
__global__ __launch_bounds__(128) void mha_out_gemm(
    const f16* __restrict__ X, const f16* __restrict__ W,
    const float* __restrict__ bias, float* __restrict__ out) {
  const int lane = threadIdx.x & 31;
  const int wave = threadIdx.x >> 5;
  const int lm   = lane & 15;
  const int hlf  = lane >> 4;
  const int m0 = blockIdx.x * 16;
  const int n0 = (blockIdx.y * 4 + wave) * 64;

  const f16* arow = X + (size_t)(m0 + lm) * Dd + 16 * hlf;
  const f16* brow = W + (size_t)(n0 + lm) * Dd + 16 * hlf;

  v8f acc[4];
#pragma unroll
  for (int t = 0; t < 4; ++t) acc[t] = zero8();

  F16Frag a, bf[4];
  a.u[0] = *(const v4u*)(arow);
  a.u[1] = *(const v4u*)(arow + 8);
#pragma unroll
  for (int t = 0; t < 4; ++t) {
    const f16* p = brow + (size_t)t * 16 * Dd;
    bf[t].u[0] = *(const v4u*)(p);
    bf[t].u[1] = *(const v4u*)(p + 8);
  }

  for (int k0 = 32; k0 < Dd; k0 += 32) {
    F16Frag a2, b2[4];
    a2.u[0] = *(const v4u*)(arow + k0);
    a2.u[1] = *(const v4u*)(arow + k0 + 8);
#pragma unroll
    for (int t = 0; t < 4; ++t) {
      const f16* p = brow + (size_t)t * 16 * Dd + k0;
      b2[t].u[0] = *(const v4u*)(p);
      b2[t].u[1] = *(const v4u*)(p + 8);
    }
#pragma unroll
    for (int t = 0; t < 4; ++t) acc[t] = wmma32(a.v, bf[t].v, acc[t]);
    a = a2;
#pragma unroll
    for (int t = 0; t < 4; ++t) bf[t] = b2[t];
  }
#pragma unroll
  for (int t = 0; t < 4; ++t) acc[t] = wmma32(a.v, bf[t].v, acc[t]);

#pragma unroll
  for (int t = 0; t < 4; ++t) {
    const int n = n0 + t * 16 + lm;
    const float bv = bias[n];
#pragma unroll
    for (int j = 0; j < 8; ++j) {
      const int row = m0 + j + 8 * hlf;
      out[(size_t)row * Dd + n] = acc[t][j] + bv;
    }
  }
}

// --------------------------------------------------------------------- launch
extern "C" void kernel_launch(void* const* d_in, const int* in_sizes, int n_in,
                              void* d_out, int out_size, void* d_ws, size_t ws_size,
                              hipStream_t stream) {
  (void)in_sizes; (void)n_in; (void)out_size; (void)ws_size;
  const float* x  = (const float*)d_in[0];
  const float* Wq = (const float*)d_in[1];
  const float* Wk = (const float*)d_in[2];
  const float* Wv = (const float*)d_in[3];
  const float* bq = (const float*)d_in[4];
  const float* bk = (const float*)d_in[5];
  const float* bv = (const float*)d_in[6];
  const float* Wo = (const float*)d_in[7];
  const float* bo = (const float*)d_in[8];

  char* ws = (char*)d_ws;
  const size_t MB = 1024 * 1024;
  f16* xh  = (f16*)(ws + 0 * MB);    // 16 MB  x in f16
  f16* Wqh = (f16*)(ws + 16 * MB);   //  6 MB  Wq|Wk|Wv f16
  f16* Woh = (f16*)(ws + 22 * MB);   //  2 MB  Wo f16
  f16* Qb  = (f16*)(ws + 24 * MB);   // 16 MB  Q  [b,h,s,hd] (pre-scaled)
  f16* Kb  = (f16*)(ws + 40 * MB);   // 16 MB  K  [b,h,s,hd]
  f16* Vtb = (f16*)(ws + 56 * MB);   // 16 MB  V^T[b,h,hd,s]
  f16* Cc  = (f16*)(ws + 72 * MB);   // 16 MB  head-concat attn f16 (88 MB total)

  mha_cvt<<<dim3((BS * Dd / 4) / 256), 256, 0, stream>>>(x, xh, BS * Dd / 4);
  mha_cvt<<<dim3((Dd * Dd / 4) / 256), 256, 0, stream>>>(Wq, Wqh, Dd * Dd / 4);
  mha_cvt<<<dim3((Dd * Dd / 4) / 256), 256, 0, stream>>>(Wk, Wqh + Dd * Dd, Dd * Dd / 4);
  mha_cvt<<<dim3((Dd * Dd / 4) / 256), 256, 0, stream>>>(Wv, Wqh + 2 * Dd * Dd, Dd * Dd / 4);
  mha_cvt<<<dim3((Dd * Dd / 4) / 256), 256, 0, stream>>>(Wo, Woh, Dd * Dd / 4);

  dim3 gg(BS / 16, 4);
  mha_qkv_gemm<<<gg, 128, 0, stream>>>(xh, Wqh,              bq, Qb, 0);
  mha_qkv_gemm<<<gg, 128, 0, stream>>>(xh, Wqh + Dd * Dd,    bk, Kb, 1);
  mha_qkv_gemm<<<gg, 128, 0, stream>>>(xh, Wqh + 2 * Dd * Dd, bv, Vtb, 2);

  // LDS: 128KB K + 128KB V^T + 8KB P scratch = 270336 B (< 320KB WGP LDS)
  mha_attn<<<dim3(Bb * Hh * 4), 256, 270336, stream>>>(Qb, Kb, Vtb, Cc);

  mha_out_gemm<<<gg, 128, 0, stream>>>(Cc, Woh, bo, (float*)d_out);
}
